// LSTMSequence_7189775254129
// MI455X (gfx1250) — compile-verified, matrix operand order checked by reading, not run on hardware
//
#include <hip/hip_runtime.h>

typedef __attribute__((ext_vector_type(16))) _Float16 v16h;
typedef __attribute__((ext_vector_type(8)))  _Float16 v8h;
typedef __attribute__((ext_vector_type(8)))  float    v8f;

#define B_TOT  256
#define T_LEN  1024
#define D_IN   8
#define HID    32
#define NGATES 128
#define NOUT   80
#define BM     16   // batch rows per workgroup (one WMMA M-tile)

__device__ __forceinline__ float fast_exp(float x) {
  return __builtin_amdgcn_exp2f(x * 1.4426950408889634f);
}
__device__ __forceinline__ float fast_sigmoid(float x) {
  return __builtin_amdgcn_rcpf(1.0f + fast_exp(-x));
}
__device__ __forceinline__ float fast_tanh(float x) {
  // tanh(x) = 2*sigmoid(2x) - 1
  return 2.0f * __builtin_amdgcn_rcpf(1.0f + fast_exp(-2.0f * x)) - 1.0f;
}

// K index held by element e (0..15) of a 16-bit A/B WMMA fragment for lane-half hi
__device__ __forceinline__ int frag_k(int e, int hi) {
  return (e >> 3) * 16 + hi * 8 + (e & 7);
}

// Load a 16x32 f16 row-major LDS tile as an A fragment: two ds_load_b128 per lane.
__device__ __forceinline__ v16h load_A_frag(const _Float16* buf, int lane) {
  const int m = lane & 15, hi = lane >> 4;
  const v8h lo = *(const v8h*)(buf + m * 32 + hi * 8);        // k = hi*8 + 0..7
  const v8h hv = *(const v8h*)(buf + m * 32 + 16 + hi * 8);   // k = 16 + hi*8 + 0..7
  v16h r;
#pragma unroll
  for (int e = 0; e < 8; ++e) { r[e] = lo[e]; r[e + 8] = hv[e]; }
  return r;
}

__global__ __launch_bounds__(256) void lstm_seq_kernel(
    const float* __restrict__ x,
    const float* __restrict__ W_ih1, const float* __restrict__ W_hh1,
    const float* __restrict__ b_ih1, const float* __restrict__ b_hh1,
    const float* __restrict__ W_ih2, const float* __restrict__ W_hh2,
    const float* __restrict__ b_ih2, const float* __restrict__ b_hh2,
    const float* __restrict__ W_lin, const float* __restrict__ b_lin,
    float* __restrict__ out)
{
  __shared__ __attribute__((aligned(16))) _Float16 A1buf[BM * 32]; // [x_t | h1[:,0:24]]
  __shared__ __attribute__((aligned(16))) _Float16 A2buf[BM * 32]; // [h1[:,24:32] | 0]
  __shared__ __attribute__((aligned(16))) _Float16 h1buf[BM * 32];
  __shared__ __attribute__((aligned(16))) _Float16 h2buf[BM * 32];
  __shared__ float gates[BM * NGATES];

  const int tid    = threadIdx.x;
  const int wave   = tid >> 5;        // 0..7
  const int lane   = tid & 31;
  const int nloc   = lane & 15;       // fragment column within N-tile
  const int hi     = lane >> 4;       // lane half
  const int wgBase = blockIdx.x * BM;

  // zero-init LDS state tiles (h=0, c=0 initial conditions; A2 upper K stays 0 forever)
  for (int i = tid; i < BM * 32; i += 256) {
    A1buf[i] = (_Float16)0.0f; A2buf[i] = (_Float16)0.0f;
    h1buf[i] = (_Float16)0.0f; h2buf[i] = (_Float16)0.0f;
  }

  // ---------- one-time per-wave weight fragments + biases ----------
  const int gn = wave * 16 + nloc;    // gate column 0..127 owned by this lane
  v16h B1, B2, Bih2, Bhh2, Blin;
#pragma unroll
  for (int e = 0; e < 16; ++e) Blin[e] = (_Float16)0.0f;
#pragma unroll
  for (int e = 0; e < 16; ++e) {
    const int k = frag_k(e, hi);
    B1[e]   = (_Float16)((k < 8) ? W_ih1[gn * D_IN + k] : W_hh1[gn * HID + (k - 8)]);
    B2[e]   = (_Float16)((k < 8) ? W_hh1[gn * HID + 24 + k] : 0.0f);
    Bih2[e] = (_Float16)W_ih2[gn * HID + k];
    Bhh2[e] = (_Float16)W_hh2[gn * HID + k];
  }
  const float bias1 = b_ih1[gn] + b_hh1[gn];
  const float bias2 = b_ih2[gn] + b_hh2[gn];
  float biasl = 0.0f;
  if (wave < 5) {
    const int on = wave * 16 + nloc;  // output column 0..79
#pragma unroll
    for (int e = 0; e < 16; ++e) Blin[e] = (_Float16)W_lin[on * HID + frag_k(e, hi)];
    biasl = b_lin[on];
  }

  // cell states in registers: thread owns (m = wave + 8*rep, j = lane)
  float c1[2] = {0.0f, 0.0f};
  float c2[2] = {0.0f, 0.0f};

  __syncthreads();

#pragma unroll 1
  for (int t = 0; t < T_LEN; ++t) {
    // stage x_t into A1buf[:, 0:8]
    if (tid < BM * D_IN) {
      const int m = tid >> 3, j = tid & 7;
      A1buf[m * 32 + j] =
          (_Float16)x[((size_t)(wgBase + m) * T_LEN + t) * D_IN + j];
    }
    __syncthreads();

    // ---------------- layer 1 gates: [x|h1] @ [W_ih1|W_hh1]^T ----------------
    {
      const v16h a1 = load_A_frag(A1buf, lane);
      const v16h a2 = load_A_frag(A2buf, lane);
      v8f c;
#pragma unroll
      for (int r = 0; r < 8; ++r) c[r] = bias1;
      c = __builtin_amdgcn_wmma_f32_16x16x32_f16(false, a1, false, B1,   (short)0, c, false, false);
      c = __builtin_amdgcn_wmma_f32_16x16x32_f16(false, a2, false, B2,   (short)0, c, false, false);
#pragma unroll
      for (int r = 0; r < 8; ++r)
        gates[(r + 8 * hi) * NGATES + wave * 16 + nloc] = c[r];
    }
    __syncthreads();

    // ---------------- layer 1 elementwise ----------------
#pragma unroll
    for (int rep = 0; rep < 2; ++rep) {
      const int m = wave + 8 * rep;
      const int j = lane;
      const float gi = gates[m * NGATES + j];
      const float gf = gates[m * NGATES + 32 + j];
      const float gg = gates[m * NGATES + 64 + j];
      const float go = gates[m * NGATES + 96 + j];
      const float cn = fast_sigmoid(gf) * c1[rep] + fast_sigmoid(gi) * fast_tanh(gg);
      c1[rep] = cn;
      const _Float16 hh = (_Float16)(fast_sigmoid(go) * fast_tanh(cn));
      h1buf[m * 32 + j] = hh;
      if (j < 24) A1buf[m * 32 + 8 + j] = hh;
      else        A2buf[m * 32 + (j - 24)] = hh;
    }
    __syncthreads();

    // ---------------- layer 2 gates: h1@W_ih2^T + h2@W_hh2^T ----------------
    {
      const v16h ah1 = load_A_frag(h1buf, lane);
      const v16h ah2 = load_A_frag(h2buf, lane);
      v8f c;
#pragma unroll
      for (int r = 0; r < 8; ++r) c[r] = bias2;
      c = __builtin_amdgcn_wmma_f32_16x16x32_f16(false, ah1, false, Bih2, (short)0, c, false, false);
      c = __builtin_amdgcn_wmma_f32_16x16x32_f16(false, ah2, false, Bhh2, (short)0, c, false, false);
#pragma unroll
      for (int r = 0; r < 8; ++r)
        gates[(r + 8 * hi) * NGATES + wave * 16 + nloc] = c[r];
    }
    __syncthreads();

    // ---------------- layer 2 elementwise ----------------
#pragma unroll
    for (int rep = 0; rep < 2; ++rep) {
      const int m = wave + 8 * rep;
      const int j = lane;
      const float gi = gates[m * NGATES + j];
      const float gf = gates[m * NGATES + 32 + j];
      const float gg = gates[m * NGATES + 64 + j];
      const float go = gates[m * NGATES + 96 + j];
      const float cn = fast_sigmoid(gf) * c2[rep] + fast_sigmoid(gi) * fast_tanh(gg);
      c2[rep] = cn;
      h2buf[m * 32 + j] = (_Float16)(fast_sigmoid(go) * fast_tanh(cn));
    }
    __syncthreads();

    // ---------------- output projection: h2 @ W_lin^T + b_lin ----------------
    if (wave < 5) {
      const v16h ah2 = load_A_frag(h2buf, lane);
      v8f c;
#pragma unroll
      for (int r = 0; r < 8; ++r) c[r] = biasl;
      c = __builtin_amdgcn_wmma_f32_16x16x32_f16(false, ah2, false, Blin, (short)0, c, false, false);
#pragma unroll
      for (int r = 0; r < 8; ++r) {
        const int m = r + 8 * hi;
        out[((size_t)(wgBase + m) * T_LEN + t) * NOUT + wave * 16 + nloc] = c[r];
      }
    }
    // next iteration's x-staging only touches A1buf[:,0:8], whose last reader
    // (this step's layer-1 WMMA) is already fenced by the barriers above.
  }
}

extern "C" void kernel_launch(void* const* d_in, const int* in_sizes, int n_in,
                              void* d_out, int out_size, void* d_ws, size_t ws_size,
                              hipStream_t stream) {
  (void)in_sizes; (void)n_in; (void)out_size; (void)d_ws; (void)ws_size;
  const float* x     = (const float*)d_in[0];
  const float* W_ih1 = (const float*)d_in[1];
  const float* W_hh1 = (const float*)d_in[2];
  const float* b_ih1 = (const float*)d_in[3];
  const float* b_hh1 = (const float*)d_in[4];
  const float* W_ih2 = (const float*)d_in[5];
  const float* W_hh2 = (const float*)d_in[6];
  const float* b_ih2 = (const float*)d_in[7];
  const float* b_hh2 = (const float*)d_in[8];
  const float* W_lin = (const float*)d_in[9];
  const float* b_lin = (const float*)d_in[10];
  float* out = (float*)d_out;

  lstm_seq_kernel<<<dim3(B_TOT / BM), dim3(256), 0, stream>>>(
      x, W_ih1, W_hh1, b_ih1, b_hh1, W_ih2, W_hh2, b_ih2, b_hh2, W_lin, b_lin, out);
}
